// Mamba_6828998001044
// MI455X (gfx1250) — matrix operand order repaired and checked by reading, not compiled, not run on hardware
//
#include <hip/hip_runtime.h>
#include <hip/hip_bf16.h>

// ---------------- problem constants ----------------
constexpr int D_MODEL = 2048;
constexpr int D_INNER = 4096;
constexpr int D_STATE = 16;
constexpr int D_CONV  = 4;
constexpr int DT_RANK = 128;
constexpr int BATCH   = 2;
constexpr int SEQLEN  = 2048;
constexpr int MROWS   = BATCH * SEQLEN;        // 4096 (flattened b,l)
constexpr int XZ_W    = 2 * D_INNER;           // 8192
constexpr int PROJ_W  = DT_RANK + 2 * D_STATE; // 160

// ---------------- vector types ----------------
typedef __attribute__((ext_vector_type(16))) __bf16 v16bf;
typedef __attribute__((ext_vector_type(8)))  __bf16 v8bf;
typedef __attribute__((ext_vector_type(8)))  float  v8f;
typedef __attribute__((ext_vector_type(4)))  unsigned int v4u;
typedef __attribute__((ext_vector_type(8)))  int v8i;
typedef __attribute__((ext_vector_type(4)))  int v4i;

static __device__ __forceinline__ v16bf combine8(v8bf a, v8bf b) {
    v16bf r;
#pragma unroll
    for (int i = 0; i < 8; ++i) { r[i] = a[i]; r[i + 8] = b[i]; }
    return r;
}

// ---------------- Tensor Data Mover: 2D bf16 tile Global -> LDS ----------------
// Builds a D# (CDNA5 ISA) for a [tile_rows x tile_k] bf16
// tile at gptr inside a row-major matrix with row stride `row_stride` elems.
// rows_rem / k_rem give the remaining tensor extent from the tile origin so the
// TDM zero-fills out-of-bounds rows (handles the ragged N=160 edge for free).
static __device__ __forceinline__ void tdm_load_bf16_2d(const void* gptr, unsigned lds_off,
                                                        int rows_rem, int k_rem,
                                                        int tile_rows, int tile_k,
                                                        int row_stride) {
    unsigned long long ga = (unsigned long long)(uintptr_t)gptr;
    v4u g0;
    g0[0] = 1u;                                   // count=1 valid descriptor
    g0[1] = lds_off;                              // lds_addr (bytes)
    g0[2] = (unsigned)ga;                         // global_addr[31:0]
    g0[3] = (unsigned)(ga >> 32) | (2u << 30);    // global_addr[56:32] | type=2
    v8i g1;
    g1[0] = 1 << 16;                              // data_size=1 (2 bytes); mask=0
    g1[1] = (int)(((unsigned)k_rem & 0xffffu) << 16);                       // tensor_dim0[15:0]
    g1[2] = (int)(((unsigned)k_rem >> 16) | (((unsigned)rows_rem & 0xffffu) << 16)); // dim0 hi | dim1 lo
    g1[3] = (int)(((unsigned)rows_rem >> 16) | ((unsigned)tile_k << 16));   // dim1 hi | tile_dim0
    g1[4] = tile_rows;                            // tile_dim1 (tile_dim2 = 0)
    g1[5] = row_stride;                           // tensor_dim0_stride[31:0]
    g1[6] = 0;                                    // stride hi | dim1_stride lo
    g1[7] = 0;
    v4i g2 = {0, 0, 0, 0};                        // group2 unused (2D tensor)
    v4i g3 = {0, 0, 0, 0};                        // group3 unused
    v8i g4 = {0, 0, 0, 0, 0, 0, 0, 0};            // extra group (clang-23 6-arg form)
    __builtin_amdgcn_tensor_load_to_lds(g0, g1, g2, g3, g4, 0);
}

// ---------------- f32 -> bf16 conversion (contiguous) ----------------
__global__ __launch_bounds__(256) void f32_to_bf16_kernel(const float* __restrict__ in,
                                                          __bf16* __restrict__ out,
                                                          long n) {
    long i = (long)blockIdx.x * blockDim.x + threadIdx.x;
    long stride = (long)gridDim.x * blockDim.x;
    for (; i < n; i += stride) out[i] = (__bf16)in[i];
}

// strided: pull dt_low (first 128 of the 160-wide proj rows) into packed bf16
__global__ __launch_bounds__(256) void dtlow_to_bf16_kernel(const float* __restrict__ proj,
                                                            __bf16* __restrict__ out) {
    int i = blockIdx.x * 256 + threadIdx.x;      // MROWS*DT_RANK
    int r = i >> 7;                               // /128
    int c = i & 127;
    out[i] = (__bf16)proj[(size_t)r * PROJ_W + c];
}

// ---------------- bf16 WMMA GEMM with TDM double-buffered staging ----------------
// C[M,N] = A[M,K] * W[N,K]^T.  Block tile 64x128, staged K-depth 64 (2 WMMA k-steps).
// 8 waves (wave32) arranged 2x4; each wave owns a 32x32 tile (2x2 WMMA tiles).
// Tile k+1 is DMA'd by the Tensor Data Mover while tile k feeds 8 v_wmma per wave.
#define BM 64
#define BN 128
#define BK 64
constexpr int ABYTES   = BM * BK * 2;             // 8 KB
constexpr int BUFBYTES = (BM + BN) * BK * 2;      // 24 KB per buffer

__global__ __launch_bounds__(256) void gemm_bf16_wmma(const __bf16* __restrict__ A,
                                                      const __bf16* __restrict__ W,
                                                      float* __restrict__ C,
                                                      int M, int N, int K) {
    // single static LDS block => base offset 0; [buf][A tile | W tile]
    __shared__ __align__(128) __bf16 smem[2][(BM + BN) * BK];   // 48 KB

    const int tid    = threadIdx.x;
    const int lane   = tid & 31;
    const int wave_s = __builtin_amdgcn_readfirstlane(tid >> 5); // uniform -> scalar branch
    const int wm     = wave_s >> 2;     // 0..1 -> 32-row slab
    const int wn     = wave_s & 3;      // 0..3 -> 32-col slab
    const int l      = lane & 15;
    const int hi     = lane >> 4;

    const int blockM = blockIdx.y * BM;
    const int blockN = blockIdx.x * BN;
    const int nk     = K / BK;

    // ---- prologue: DMA tile 0 into buffer 0
    if (wave_s == 0)
        tdm_load_bf16_2d(A + (size_t)blockM * K, 0u, M - blockM, K, BM, BK, K);
    else if (wave_s == 1)
        tdm_load_bf16_2d(W + (size_t)blockN * K, (unsigned)ABYTES, N - blockN, K, BN, BK, K);
    if (wave_s < 2) __builtin_amdgcn_s_wait_tensorcnt(0);
    __syncthreads();

    v8f acc[2][2] = {};

    for (int ki = 0; ki < nk; ++ki) {
        const int buf = ki & 1;

        // ---- async prefetch of tile ki+1 into the other buffer
        if (ki + 1 < nk) {
            int k0n = (ki + 1) * BK;
            if (wave_s == 0)
                tdm_load_bf16_2d(A + (size_t)blockM * K + k0n,
                                 (unsigned)((buf ^ 1) * BUFBYTES),
                                 M - blockM, K - k0n, BM, BK, K);
            else if (wave_s == 1)
                tdm_load_bf16_2d(W + (size_t)blockN * K + k0n,
                                 (unsigned)((buf ^ 1) * BUFBYTES + ABYTES),
                                 N - blockN, K - k0n, BN, BK, K);
        }

        const __bf16* As = &smem[buf][0];
        const __bf16* Ws = &smem[buf][BM * BK];

        // ---- 2 WMMA k-steps per staged tile (8 v_wmma per wave per sync interval)
#pragma unroll
        for (int kk = 0; kk < BK; kk += 32) {
            // fragments (ISA 7.12.2 16-bit layouts)
            v16bf afrag[2], bfrag[2];
#pragma unroll
            for (int i = 0; i < 2; ++i) {
                int m = wm * 32 + i * 16 + l;
                v8bf lo = *(const v8bf*)&As[m * BK + kk + hi * 8];        // K = kk+hi*8 ..
                v8bf hv = *(const v8bf*)&As[m * BK + kk + 16 + hi * 8];   // K = kk+16+hi*8 ..
                afrag[i] = combine8(lo, hv);
            }
#pragma unroll
            for (int j = 0; j < 2; ++j) {
                int n = wn * 32 + j * 16 + l;
                v8bf blo = *(const v8bf*)&Ws[n * BK + kk + hi * 8];       // K = kk+hi*8 ..
                v8bf bhi = *(const v8bf*)&Ws[n * BK + kk + 16 + hi * 8];  // K = kk+16+hi*8 ..
                bfrag[j] = combine8(blo, bhi);
            }

#pragma unroll
            for (int i = 0; i < 2; ++i)
#pragma unroll
                for (int j = 0; j < 2; ++j)
                    acc[i][j] = __builtin_amdgcn_wmma_f32_16x16x32_bf16(
                        false, afrag[i], false, bfrag[j], (short)0, acc[i][j], false, false);
        }

        // ---- publish next tile: DMA issuers drain TENSORcnt, then all sync
        if (ki + 1 < nk && wave_s < 2) __builtin_amdgcn_s_wait_tensorcnt(0);
        __syncthreads();
    }

    // ---- epilogue: C/D layout -> lane l = N col, VGPR r = M row hi*8+r
#pragma unroll
    for (int i = 0; i < 2; ++i) {
#pragma unroll
        for (int j = 0; j < 2; ++j) {
            int gn = blockN + wn * 32 + j * 16 + l;
#pragma unroll
            for (int r = 0; r < 8; ++r) {
                int gm = blockM + wm * 32 + i * 16 + hi * 8 + r;
                if (gm < M && gn < N) C[(size_t)gm * N + gn] = acc[i][j][r];
            }
        }
    }
}

// ---------------- depthwise causal conv (k=4) + bias + SiLU ----------------
__global__ __launch_bounds__(256) void conv_silu_kernel(const float* __restrict__ xz,
                                                        const float* __restrict__ conv_w,
                                                        const float* __restrict__ conv_b,
                                                        float* __restrict__ xt,
                                                        __bf16* __restrict__ xtb) {
    size_t i = (size_t)blockIdx.x * 256 + threadIdx.x;   // < MROWS*D_INNER
    int d = (int)(i % D_INNER);
    size_t bl = i / D_INNER;
    int lpos = (int)(bl % SEQLEN);
    int b = (int)(bl / SEQLEN);

    float acc = conv_b[d];
#pragma unroll
    for (int j = 0; j < D_CONV; ++j) {
        int ls = lpos - (D_CONV - 1) + j;
        if (ls >= 0)
            acc += conv_w[d * D_CONV + j] * xz[((size_t)b * SEQLEN + ls) * XZ_W + d];
    }
    float s = acc / (1.f + __expf(-acc));   // SiLU
    xt[i] = s;
    xtb[i] = (__bf16)s;
}

// ---------------- fused selective scan ----------------
__global__ __launch_bounds__(256) void scan_kernel(const float* __restrict__ dt_raw, // (MROWS, D_INNER)
                                                   const float* __restrict__ dt_b,   // (D_INNER)
                                                   const float* __restrict__ xt,     // (MROWS, D_INNER)
                                                   const float* __restrict__ proj,   // (MROWS, 160)
                                                   const float* __restrict__ A_log,  // (D_INNER, 16)
                                                   const float* __restrict__ Dvec,   // (D_INNER)
                                                   const float* __restrict__ xz,     // z at +4096
                                                   __bf16* __restrict__ ybf) {       // (MROWS, D_INNER)
    const int b    = blockIdx.x >> 4;            // 16 blocks per batch
    const int dblk = blockIdx.x & 15;
    const int d    = dblk * 256 + threadIdx.x;

    float Aneg[D_STATE];
#pragma unroll
    for (int n = 0; n < D_STATE; ++n) Aneg[n] = -__expf(A_log[(size_t)d * D_STATE + n]);
    float h[D_STATE];
#pragma unroll
    for (int n = 0; n < D_STATE; ++n) h[n] = 0.f;

    const float dtb = dt_b[d];
    const float Dd  = Dvec[d];

    __shared__ float sB[D_STATE];
    __shared__ float sC[D_STATE];

    for (int lpos = 0; lpos < SEQLEN; ++lpos) {
        size_t m = (size_t)b * SEQLEN + lpos;
        if (threadIdx.x < D_STATE)
            sB[threadIdx.x] = proj[m * PROJ_W + DT_RANK + threadIdx.x];
        else if (threadIdx.x < 2 * D_STATE)
            sC[threadIdx.x - D_STATE] = proj[m * PROJ_W + DT_RANK + D_STATE + (threadIdx.x - D_STATE)];
        __syncthreads();

        float dtv = dt_raw[m * D_INNER + d] + dtb;
        dtv = (dtv > 20.f) ? dtv : log1pf(__expf(dtv));  // softplus
        float u  = xt[m * D_INNER + d];
        float du = dtv * u;

        float y = 0.f;
#pragma unroll
        for (int n = 0; n < D_STATE; ++n) {
            float dA = __expf(dtv * Aneg[n]);
            h[n] = dA * h[n] + du * sB[n];
            y += h[n] * sC[n];
        }
        y += u * Dd;

        float z = xz[m * XZ_W + D_INNER + d];
        y *= z / (1.f + __expf(-z));                      // * silu(z)

        ybf[m * D_INNER + d] = (__bf16)y;
        __syncthreads();
    }
}

// ---------------- host-side orchestration ----------------
static inline size_t align256(size_t x) { return (x + 255) & ~(size_t)255; }

extern "C" void kernel_launch(void* const* d_in, const int* in_sizes, int n_in,
                              void* d_out, int out_size, void* d_ws, size_t ws_size,
                              hipStream_t stream) {
    const float* hs        = (const float*)d_in[0];  // (2,2048,2048)
    const float* in_proj_w = (const float*)d_in[1];  // (8192,2048)
    const float* conv_w    = (const float*)d_in[2];  // (4096,4)
    const float* conv_b    = (const float*)d_in[3];  // (4096)
    const float* x_proj_w  = (const float*)d_in[4];  // (160,4096)
    const float* dt_proj_w = (const float*)d_in[5];  // (4096,128)
    const float* dt_proj_b = (const float*)d_in[6];  // (4096)
    const float* A_log     = (const float*)d_in[7];  // (4096,16)
    const float* Dvec      = (const float*)d_in[8];  // (4096)
    const float* out_proj_w= (const float*)d_in[9];  // (2048,4096)
    float* out = (float*)d_out;                       // (2,2048,2048) f32

    char* ws = (char*)d_ws;
    size_t off = 0;
    auto carve = [&](size_t bytes) { char* p = ws + off; off = align256(off + bytes); return p; };

    __bf16* hsb   = (__bf16*)carve((size_t)MROWS * D_MODEL * 2);
    __bf16* winb  = (__bf16*)carve((size_t)XZ_W * D_MODEL * 2);
    float*  xz    = (float*) carve((size_t)MROWS * XZ_W * 4);
    float*  xt    = (float*) carve((size_t)MROWS * D_INNER * 4);
    __bf16* xtb   = (__bf16*)carve((size_t)MROWS * D_INNER * 2);
    __bf16* wxb   = (__bf16*)carve((size_t)PROJ_W * D_INNER * 2);
    float*  proj  = (float*) carve((size_t)MROWS * PROJ_W * 4);
    __bf16* dtlb  = (__bf16*)carve((size_t)MROWS * DT_RANK * 2);
    __bf16* wdtb  = (__bf16*)carve((size_t)D_INNER * DT_RANK * 2);
    float*  dtraw = (float*) carve((size_t)MROWS * D_INNER * 4);
    __bf16* ybf   = (__bf16*)carve((size_t)MROWS * D_INNER * 2);
    __bf16* woutb = (__bf16*)carve((size_t)D_MODEL * D_INNER * 2);
    (void)ws_size; (void)n_in; (void)in_sizes; (void)out_size;

    // 1) bf16 conversions (streaming)
    auto conv_launch = [&](const float* src, __bf16* dst, long n) {
        int blocks = (int)((n + 255) / 256);
        if (blocks > 4096) blocks = 4096;
        f32_to_bf16_kernel<<<blocks, 256, 0, stream>>>(src, dst, n);
    };
    conv_launch(hs,        hsb,   (long)MROWS * D_MODEL);
    conv_launch(in_proj_w, winb,  (long)XZ_W * D_MODEL);
    conv_launch(x_proj_w,  wxb,   (long)PROJ_W * D_INNER);
    conv_launch(dt_proj_w, wdtb,  (long)D_INNER * DT_RANK);
    conv_launch(out_proj_w,woutb, (long)D_MODEL * D_INNER);

    // 2) in_proj GEMM: xz[M,8192] = hs[M,2048] * in_proj_w[8192,2048]^T
    {
        dim3 g((XZ_W + BN - 1) / BN, (MROWS + BM - 1) / BM);
        gemm_bf16_wmma<<<g, 256, 0, stream>>>(hsb, winb, xz, MROWS, XZ_W, D_MODEL);
    }

    // 3) depthwise conv + SiLU -> xt (f32 + bf16)
    {
        size_t total = (size_t)MROWS * D_INNER;
        conv_silu_kernel<<<(unsigned)(total / 256), 256, 0, stream>>>(xz, conv_w, conv_b, xt, xtb);
    }

    // 4) x_proj GEMM: proj[M,160] = xt[M,4096] * x_proj_w[160,4096]^T
    {
        dim3 g((PROJ_W + BN - 1) / BN, (MROWS + BM - 1) / BM);
        gemm_bf16_wmma<<<g, 256, 0, stream>>>(xtb, wxb, proj, MROWS, PROJ_W, D_INNER);
    }

    // 5) dt_low -> bf16 (strided gather from proj)
    {
        int total = MROWS * DT_RANK;
        dtlow_to_bf16_kernel<<<total / 256, 256, 0, stream>>>(proj, dtlb);
    }

    // 6) dt GEMM: dtraw[M,4096] = dt_low[M,128] * dt_proj_w[4096,128]^T
    {
        dim3 g((D_INNER + BN - 1) / BN, (MROWS + BM - 1) / BM);
        gemm_bf16_wmma<<<g, 256, 0, stream>>>(dtlb, wdtb, dtraw, MROWS, D_INNER, DT_RANK);
    }

    // 7) fused selective scan (+softplus, +u*D, *silu(z)) -> y bf16
    scan_kernel<<<BATCH * (D_INNER / 256), 256, 0, stream>>>(
        dtraw, dt_proj_b, xt, proj, A_log, Dvec, xz, ybf);

    // 8) out_proj GEMM: out[M,2048] = y[M,4096] * out_proj_w[2048,4096]^T
    {
        dim3 g((D_MODEL + BN - 1) / BN, (MROWS + BM - 1) / BM);
        gemm_bf16_wmma<<<g, 256, 0, stream>>>(ybf, woutb, out, MROWS, D_MODEL, D_INNER);
    }
}
